// NonEquiNet_65627100283289
// MI455X (gfx1250) — matrix operand order repaired and checked by reading, not compiled, not run on hardware
//
#include <hip/hip_runtime.h>

typedef unsigned int   u32;
typedef unsigned short u16;
typedef __attribute__((ext_vector_type(16))) __bf16 v16bf;
typedef __attribute__((ext_vector_type(2)))  __bf16 v2bf;
typedef __attribute__((ext_vector_type(8)))  float  v8f;
typedef __attribute__((ext_vector_type(8)))  u32    v8u;

// ---------- fp32 pair -> packed bf16 dword via native converts ----------
__device__ __forceinline__ u32 pk2(float a, float b) {
  v2bf t;
  t[0] = (__bf16)a;
  t[1] = (__bf16)b;
  return __builtin_bit_cast(u32, t);
}

// ---------- zero fill ----------
__global__ void zero_kernel(float* __restrict__ p, int n) {
  int i = blockIdx.x * blockDim.x + threadIdx.x;
  if (i < n) p[i] = 0.0f;
}

// ---------- pack fp32 W[K,N] into bf16 A-fragment-native layout ----------
// The MLP GEMM is computed as D = W^T x X^T, so the weight matrix is the WMMA
// A operand. packed 16-bit A 16x32 layout (contiguous k per lane half):
//   lane L: M = t*16 + (L&15)   (output feature)
//   VGPR v   : k = s*32 + (L>=16?16:0) + 2v
// dword gid layout: (((t*Ksteps + s)*32 + L)*8 + v). k >= Kreal rows are zero
// (pads block0's K=8 input up to K=32).
__global__ void pack_weights_kernel(const float* __restrict__ W, u32* __restrict__ P,
                                    int Kreal, int Kpad, int Nw, int total) {
  int gid = blockIdx.x * blockDim.x + threadIdx.x;
  if (gid >= total) return;
  int v = gid & 7;
  int L = (gid >> 3) & 31;
  int f = gid >> 8;
  int ks = Kpad >> 5;
  int s = f % ks;
  int t = f / ks;
  int n = t * 16 + (L & 15);
  int kbase = s * 32 + ((L >> 4) << 4);
  int k = kbase + 2 * v;
  float lo = (k     < Kreal) ? W[(size_t)k       * Nw + n] : 0.0f;
  float hi = (k + 1 < Kreal) ? W[(size_t)(k + 1) * Nw + n] : 0.0f;
  P[gid] = pk2(lo, hi);
}

// ---------- one MLP layer on a 32-edge (2 N-tile) slab, wave-private ----------
// act: LDS bf16 activations, 32 edge-rows x K (row stride STRIDE u16, 32B aligned)
// wp : packed weights (A-fragment order) in global; each feeds 2 WMMAs.
// D(NW x 32) = W^T(NW x K) * X^T(K x 32) + bias; relu optional;
// FINAL scatters fp32 atomics into gout[dst[e]*NW + feature].
template <int K, int STRIDE, int NW, bool RELU, bool FINAL>
__device__ __forceinline__ void mlp_layer(
    const u16* __restrict__ act, const u32* __restrict__ wp,
    const float* __restrict__ bias,
    u16* __restrict__ actout, int sout,
    float* __restrict__ gout, const int* __restrict__ dstIdx,
    int ebase, int E, int lane) {
  constexpr int KS = K / 32;
  const int col = lane & 15;   // edge column within an N-tile / feature within A
  const int hi  = lane >> 4;

  // B fragments (activations) for both 16-edge N-tiles, cached in VGPRs.
  // B layout: lanes 0-15 hold K=0..15 (pairs per VGPR), lanes 16-31 hold K=16..31
  // -> 8 contiguous dwords per lane from row-major [edge][k] LDS (2x ds_load_b128).
  const v8u* actv = (const v8u*)act;
  v16bf bfr0[KS], bfr1[KS];
  const int b0 = col * (STRIDE / 16) + hi;          // v8u units (32B)
  const int b1 = (col + 16) * (STRIDE / 16) + hi;
#pragma unroll
  for (int s = 0; s < KS; ++s) {
    bfr0[s] = __builtin_bit_cast(v16bf, actv[b0 + s * 2]);
    bfr1[s] = __builtin_bit_cast(v16bf, actv[b1 + s * 2]);
  }

  int dn0 = 0, dn1 = 0;
  if constexpr (FINAL) {
    int e0 = ebase + col;
    int e1 = e0 + 16;
    dn0 = (e0 < E) ? dstIdx[e0] : -1;
    dn1 = (e1 < E) ? dstIdx[e1] : -1;
  }

  const v8u* wv  = (const v8u*)wp;
  u32*       aov = (u32*)actout;
#pragma unroll
  for (int t = 0; t < NW / 16; ++t) {
    v8f c0 = {};
    v8f c1 = {};
#pragma unroll
    for (int s = 0; s < KS; ++s) {
      v8u ww = wv[(t * KS + s) * 32 + lane];
      v16bf wfr = __builtin_bit_cast(v16bf, ww);
      c0 = __builtin_amdgcn_wmma_f32_16x16x32_bf16(false, wfr, false, bfr0[s],
                                                   (short)0, c0, false, false);
      c1 = __builtin_amdgcn_wmma_f32_16x16x32_bf16(false, wfr, false, bfr1[s],
                                                   (short)0, c1, false, false);
    }
    // lane owns features fbase..fbase+7 of edges col (c0) and col+16 (c1)
    const int fbase = t * 16 + hi * 8;
    if constexpr (FINAL) {
#pragma unroll
      for (int v = 0; v < 8; v += 2) {
        float2 bb = *(const float2*)(bias + fbase + v);
        float p0 = c0[v] + bb.x, p1 = c0[v + 1] + bb.y;
        float q0 = c1[v] + bb.x, q1 = c1[v + 1] + bb.y;
        if (dn0 >= 0) {
          atomicAdd(&gout[(size_t)dn0 * NW + fbase + v],     p0);
          atomicAdd(&gout[(size_t)dn0 * NW + fbase + v + 1], p1);
        }
        if (dn1 >= 0) {
          atomicAdd(&gout[(size_t)dn1 * NW + fbase + v],     q0);
          atomicAdd(&gout[(size_t)dn1 * NW + fbase + v + 1], q1);
        }
      }
    } else {
#pragma unroll
      for (int v = 0; v < 8; v += 2) {
        float2 bb = *(const float2*)(bias + fbase + v);
        float p0 = c0[v] + bb.x, p1 = c0[v + 1] + bb.y;
        float q0 = c1[v] + bb.x, q1 = c1[v + 1] + bb.y;
        if (RELU) {
          p0 = fmaxf(p0, 0.0f); p1 = fmaxf(p1, 0.0f);
          q0 = fmaxf(q0, 0.0f); q1 = fmaxf(q1, 0.0f);
        }
        // contiguous feature pair of one edge -> single packed dword store
        aov[col * (sout / 2) + (fbase + v) / 2]        = pk2(p0, p1);
        aov[(col + 16) * (sout / 2) + (fbase + v) / 2] = pk2(q0, q1);
      }
    }
  }
}

// ---------- EdgeConv block: gather -> 3-layer MLP -> atomic segment-sum ----------
// DINF: feature width of xin (4 or 128); K0: padded concat width (32 or 256);
// NOUT: 128 or 16. 8 waves per block, 32 edges per wave (weights reused 2x).
template <int DINF, int K0, int NOUT>
__global__ __launch_bounds__(256) void edgeconv_kernel(
    const float* __restrict__ xin,
    const int* __restrict__ srcIdx, const int* __restrict__ dstIdx,
    const u32* __restrict__ wp0, const float* __restrict__ bias0,
    const u32* __restrict__ wp1, const float* __restrict__ bias1,
    const u32* __restrict__ wp2, const float* __restrict__ bias2,
    float* __restrict__ out, int E) {
  constexpr int SA = (K0 > 128) ? K0 : 128;  // bufA row stride (u16)
  extern __shared__ u16 smem[];
  const int lane = threadIdx.x & 31;
  const int wave = threadIdx.x >> 5;
  u16* bufA = smem + wave * 32 * (SA + 128);
  u16* bufB = bufA + 32 * SA;

  const int tile  = blockIdx.x * 8 + wave;
  const int ebase = tile * 32;
  if (ebase >= E) return;

  // gather cat(x_i, x_j - x_i) -> bufA (bf16 row-major [edge][k]), zero-pad to K0
  if (DINF == 4) {
    int e = ebase + lane; if (e >= E) e = E - 1;
    int sN = srcIdx[e], dN = dstIdx[e];
    const float* xi = xin + (size_t)dN * 4;
    const float* xj = xin + (size_t)sN * 4;
    float a0 = xi[0], a1 = xi[1], a2 = xi[2], a3 = xi[3];
    float b0 = xj[0], b1 = xj[1], b2 = xj[2], b3 = xj[3];
    u32* rowp = (u32*)(bufA + lane * SA);
    rowp[0] = pk2(a0, a1);           rowp[1] = pk2(a2, a3);
    rowp[2] = pk2(b0 - a0, b1 - a1); rowp[3] = pk2(b2 - a2, b3 - a3);
#pragma unroll
    for (int i = 4; i < 16; ++i) rowp[i] = 0u;   // pad K 8..31
  } else {
    int e = ebase + lane; if (e >= E) e = E - 1;
    int sN = srcIdx[e], dN = dstIdx[e];
    const float4* xd = (const float4*)(xin + (size_t)dN * DINF);
    const float4* xs = (const float4*)(xin + (size_t)sN * DINF);
    u32* rowp = (u32*)(bufA + lane * SA);
#pragma unroll 4
    for (int j = 0; j < DINF / 4; ++j) {
      float4 a = xd[j];
      float4 s4 = xs[j];
      rowp[j * 2 + 0] = pk2(a.x, a.y);                           // cols [0, DINF)  : x_i
      rowp[j * 2 + 1] = pk2(a.z, a.w);
      rowp[DINF / 2 + j * 2 + 0] = pk2(s4.x - a.x, s4.y - a.y);  // cols [DINF,2*DINF): x_j - x_i
      rowp[DINF / 2 + j * 2 + 1] = pk2(s4.z - a.z, s4.w - a.w);
    }
  }

  // 3-layer MLP, ping-pong LDS buffers (wave-private, no barriers)
  mlp_layer<K0, SA, 128, true, false>(bufA, wp0, bias0, bufB, 128,
                                      nullptr, nullptr, 0, 0, lane);
  mlp_layer<128, 128, 128, true, false>(bufB, wp1, bias1, bufA, SA,
                                        nullptr, nullptr, 0, 0, lane);
  mlp_layer<128, SA, NOUT, false, true>(bufA, wp2, bias2, nullptr, 0,
                                        out, dstIdx, ebase, E, lane);
}

// =====================================================================
extern "C" void kernel_launch(void* const* d_in, const int* in_sizes, int n_in,
                              void* d_out, int out_size, void* d_ws, size_t ws_size,
                              hipStream_t stream) {
  const float* x  = (const float*)d_in[0];
  const int*   ei = (const int*)d_in[2];
  const int E = in_sizes[2] / 2;
  const int N = in_sizes[0] / 4;
  const int* src = ei;
  const int* dst = ei + E;

  const float* W[12];
  const float* B[12];
  for (int i = 0; i < 12; ++i) {
    W[i] = (const float*)d_in[4 + i * 2];
    B[i] = (const float*)d_in[5 + i * 2];
  }

  // workspace layout
  float* xb0 = (float*)d_ws;                       // N*128 f32
  float* xb1 = xb0 + (size_t)N * 128;              // N*128 f32
  u32*   wpk = (u32*)(xb1 + (size_t)N * 128);      // packed bf16 weights

  static const int kreal[12] = {8,128,128, 256,128,128, 256,128,128, 256,128,128};
  static const int kpad [12] = {32,128,128, 256,128,128, 256,128,128, 256,128,128};
  static const int nwid [12] = {128,128,128, 128,128,128, 128,128,128, 128,128,16};

  u32* P[12];
  size_t off = 0;
  for (int i = 0; i < 12; ++i) {
    P[i] = wpk + off;
    off += (size_t)kpad[i] * nwid[i] / 2;  // dwords
  }

  // pack all weights (cheap; ~110K dwords total)
  for (int i = 0; i < 12; ++i) {
    int total = kpad[i] * nwid[i] / 2;
    pack_weights_kernel<<<(total + 255) / 256, 256, 0, stream>>>(
        W[i], P[i], kreal[i], kpad[i], nwid[i], total);
  }

  const int tiles = (E + 31) / 32;           // 32 edges per wave
  const int grid  = (tiles + 7) / 8;         // 8 waves per workgroup
  const size_t lds0   = 8u * 32 * (128 + 128) * sizeof(u16);  // 128 KB
  const size_t ldsMid = 8u * 32 * (256 + 128) * sizeof(u16);  // 192 KB

  // block 0: x[N,4] -> xb0[N,128]
  zero_kernel<<<((size_t)N * 128 + 255) / 256, 256, 0, stream>>>(xb0, N * 128);
  edgeconv_kernel<4, 32, 128><<<grid, 256, lds0, stream>>>(
      x, src, dst, P[0], B[0], P[1], B[1], P[2], B[2], xb0, E);

  // block 1: xb0 -> xb1
  zero_kernel<<<((size_t)N * 128 + 255) / 256, 256, 0, stream>>>(xb1, N * 128);
  edgeconv_kernel<128, 256, 128><<<grid, 256, ldsMid, stream>>>(
      xb0, src, dst, P[3], B[3], P[4], B[4], P[5], B[5], xb1, E);

  // block 2: xb1 -> xb0
  zero_kernel<<<((size_t)N * 128 + 255) / 256, 256, 0, stream>>>(xb0, N * 128);
  edgeconv_kernel<128, 256, 128><<<grid, 256, ldsMid, stream>>>(
      xb1, src, dst, P[6], B[6], P[7], B[7], P[8], B[8], xb0, E);

  // block 3: xb0 -> d_out[N,16]
  float* outp = (float*)d_out;
  zero_kernel<<<((size_t)N * 16 + 255) / 256, 256, 0, stream>>>(outp, N * 16);
  edgeconv_kernel<128, 256, 16><<<grid, 256, ldsMid, stream>>>(
      xb0, src, dst, P[9], B[9], P[10], B[10], P[11], B[11], outp, E);
}
